// SelfAttention_77171972374978
// MI455X (gfx1250) — matrix operand order repaired and checked by reading, not compiled, not run on hardware
//
#include <hip/hip_runtime.h>

typedef __attribute__((ext_vector_type(16))) _Float16 v16h;
typedef __attribute__((ext_vector_type(8)))  _Float16 v8h;
typedef __attribute__((ext_vector_type(8)))  float    v8f;
typedef __attribute__((ext_vector_type(4)))  float    v4f;

#define B_ 4
#define S_ 2048
#define D_ 1024

union V16H { v16h v; v8h h[2]; };

__device__ __forceinline__ v8f zero_v8f() {
  v8f z = {0.f,0.f,0.f,0.f,0.f,0.f,0.f,0.f};
  return z;
}

// ---------------------------------------------------------------------------
// CDNA5 async global->LDS copy (ASYNCcnt-tracked), guarded by __has_builtin.
// Builtin signature (from hipcc diagnostic): (AS1 v4i*, AS3 v4i*, imm int, imm int)
// where v4i is a GCC-style vector: int __attribute__((vector_size(16))).
#if __has_builtin(__builtin_amdgcn_global_load_async_to_lds_b128)
#define HAVE_ASYNC_LDS 1
typedef int async_v4i __attribute__((vector_size(16)));
typedef __attribute__((address_space(1))) async_v4i* async_gptr;
typedef __attribute__((address_space(3))) async_v4i* async_lptr;
__device__ __forceinline__ void async_copy16(const _Float16* g, _Float16* l) {
  __builtin_amdgcn_global_load_async_to_lds_b128(
      (async_gptr)g, (async_lptr)l, 0, 0);
}
__device__ __forceinline__ void wait_async_le2() {
#if __has_builtin(__builtin_amdgcn_s_wait_asynccnt)
  __builtin_amdgcn_s_wait_asynccnt(2);
#else
  asm volatile("s_wait_asynccnt 0x2" ::: "memory");
#endif
}
__device__ __forceinline__ void wait_async_0() {
#if __has_builtin(__builtin_amdgcn_s_wait_asynccnt)
  __builtin_amdgcn_s_wait_asynccnt(0);
#else
  asm volatile("s_wait_asynccnt 0x0" ::: "memory");
#endif
}
#else
#define HAVE_ASYNC_LDS 0
#endif

// ---------------------------------------------------------------------------
// WMMA fragment loaders (CDNA5 wave32 layouts, 16x16x32 f16 -> f32)
// A: 16x32 (MxK), row-major source, row stride `stride` (in halves).
//   lane m = l&15; VGPR0-3 = K {ko..ko+7}, VGPR4-7 = K {ko+16..ko+23},
//   ko = (l<16) ? 0 : 8.
__device__ __forceinline__ v16h load_frag_a(const _Float16* base, int stride, int lane) {
  const int m  = lane & 15;
  const int ko = (lane & 16) ? 8 : 0;
  const _Float16* p = base + m * stride + ko;
  V16H u;
  u.h[0] = *(const v8h*)(p);
  u.h[1] = *(const v8h*)(p + 16);
  return u.v;
}

// B: 32x16 (KxN) where the source is row-major [N][K] (row = output column n),
// row stride `stride` (halves). lane n = l&15; VGPR0-3 = K {ko..ko+7},
// VGPR4-7 = K {ko+16..ko+23}, ko = (l<16) ? 0 : 8.
__device__ __forceinline__ v16h load_frag_b(const _Float16* base, int stride, int lane) {
  const int n  = lane & 15;
  const int ko = (lane & 16) ? 8 : 0;
  const _Float16* p = base + n * stride + ko;
  V16H u;
  u.h[0] = *(const v8h*)(p);
  u.h[1] = *(const v8h*)(p + 16);
  return u.v;
}

__device__ __forceinline__ v8f wmma_f16(v16h a, v16h b, v8f c) {
  return __builtin_amdgcn_wmma_f32_16x16x32_f16(false, a, false, b, (short)0, c, false, false);
}

__device__ __forceinline__ void cvt_store8(const float* __restrict__ src,
                                           _Float16* __restrict__ dst) {
  v4f a = *(const v4f*)src;
  v4f b = *(const v4f*)(src + 4);
  v8h h;
  h[0] = (_Float16)a[0]; h[1] = (_Float16)a[1];
  h[2] = (_Float16)a[2]; h[3] = (_Float16)a[3];
  h[4] = (_Float16)b[0]; h[5] = (_Float16)b[1];
  h[6] = (_Float16)b[2]; h[7] = (_Float16)b[3];
  *(v8h*)dst = h;
}

// ---------------------------------------------------------------------------
// Kernel 1: Q/K projections.  C[m,n] = sum_k x[m,k]*W[n,k] + bias[n] -> f16.
// grid: (M/128, N/128, 2)  z=0 -> (Wq,bq,Qh), z=1 -> (Wk,bk,Kh).  block: 256.
__global__ __launch_bounds__(256) void proj_kernel(
    const float* __restrict__ x,
    const float* __restrict__ Wq, const float* __restrict__ bq,
    const float* __restrict__ Wk, const float* __restrict__ bk,
    _Float16* __restrict__ Qh, _Float16* __restrict__ Kh) {

  const float* W    = blockIdx.z ? Wk : Wq;
  const float* bias = blockIdx.z ? bk : bq;
  _Float16*   outp  = blockIdx.z ? Kh : Qh;

  __shared__ _Float16 As[128][40];   // 128 rows of x, 32 K (+pad)
  __shared__ _Float16 Bs[128][40];   // 128 rows of W (= output cols), 32 K

  const int m0 = blockIdx.x * 128;
  const int n0 = blockIdx.y * 128;
  const int tid  = threadIdx.x;
  const int lane = tid & 31;
  const int wave = tid >> 5;
  const int wm = wave >> 2;          // 0..1  -> 64-row wave tile
  const int wn = wave & 3;           // 0..3  -> 32-col wave tile

  v8f acc[4][2];
#pragma unroll
  for (int i = 0; i < 4; i++)
#pragma unroll
    for (int j = 0; j < 2; j++) acc[i][j] = zero_v8f();

  const int r  = tid >> 1;
  const int hf = (tid & 1) * 16;

  for (int k0 = 0; k0 < D_; k0 += 32) {
    __syncthreads();
    // stage x and W chunks, f32 -> f16
    {
      const float* srcA = x + (size_t)(m0 + r) * D_ + k0 + hf;
      cvt_store8(srcA,     &As[r][hf]);
      cvt_store8(srcA + 8, &As[r][hf + 8]);
      const float* srcB = W + (size_t)(n0 + r) * D_ + k0 + hf;
      cvt_store8(srcB,     &Bs[r][hf]);
      cvt_store8(srcB + 8, &Bs[r][hf + 8]);
    }
    __syncthreads();

    v16h bf0 = load_frag_b(&Bs[wn * 32][0],      40, lane);
    v16h bf1 = load_frag_b(&Bs[wn * 32 + 16][0], 40, lane);
#pragma unroll
    for (int i = 0; i < 4; i++) {
      v16h af = load_frag_a(&As[wm * 64 + i * 16][0], 40, lane);
      acc[i][0] = wmma_f16(af, bf0, acc[i][0]);
      acc[i][1] = wmma_f16(af, bf1, acc[i][1]);
    }
  }

  // epilogue: + bias, convert, store f16
  const int ncol = lane & 15;
  const int hi8  = (lane & 16) ? 8 : 0;
#pragma unroll
  for (int j = 0; j < 2; j++) {
    const int n  = n0 + wn * 32 + j * 16 + ncol;
    const float bv = bias[n];
#pragma unroll
    for (int i = 0; i < 4; i++) {
      const int mb = m0 + wm * 64 + i * 16 + hi8;
#pragma unroll
      for (int v = 0; v < 8; v++)
        outp[(size_t)(mb + v) * D_ + n] = (_Float16)(acc[i][j][v] + bv);
    }
  }
}

// ---------------------------------------------------------------------------
// Kernel 2: Xt[b][d][s] = f16(x[b][s][d]).  grid (S/32, D/32, B), block (32,8)
__global__ void transpose_kernel(const float* __restrict__ x,
                                 _Float16* __restrict__ Xt) {
  __shared__ float tile[32][33];
  const int b  = blockIdx.z;
  const int s0 = blockIdx.x * 32;
  const int d0 = blockIdx.y * 32;
  const int tx = threadIdx.x, ty = threadIdx.y;
#pragma unroll
  for (int i = ty; i < 32; i += 8)
    tile[i][tx] = x[((size_t)(b * S_ + s0 + i)) * D_ + d0 + tx];
  __syncthreads();
#pragma unroll
  for (int i = ty; i < 32; i += 8)
    Xt[((size_t)(b * D_ + d0 + i)) * S_ + s0 + tx] = (_Float16)tile[tx][i];
}

// ---------------------------------------------------------------------------
// Kernel 3: flash attention. grid (S/32, B), block 256 (8 waves).
// Wave w: 16-key score tile per 128-key block, and output columns [128w,128w+128).
// Score K-tiles are prefetched global->LDS with the CDNA5 async path
// (per-wave double buffer, ASYNCcnt-tracked, no barriers needed).
__global__ __launch_bounds__(256) void attn_kernel(
    const _Float16* __restrict__ Qh,
    const _Float16* __restrict__ Kh,
    const _Float16* __restrict__ Xt,
    float* __restrict__ out) {

  __shared__ _Float16 Qs[32][1032];      // Q tile (padded)
  __shared__ float    Sl[32][132];       // raw scores for current 128 keys
  __shared__ _Float16 Pl[32][136];       // softmaxed probs (f16)
#if HAVE_ASYNC_LDS
  __shared__ _Float16 Ksw[8][2][16][40]; // per-wave K chunk double buffer
#endif
  __shared__ float    m_st[32], l_st[32], sc_st[32];

  const int b    = blockIdx.y;
  const int q0   = blockIdx.x * 32;
  const int tid  = threadIdx.x;
  const int lane = tid & 31;
  const int w    = tid >> 5;

  // load Q tile (32 x 1024 f16) into LDS
  {
    const _Float16* src = Qh + ((size_t)(b * S_ + q0)) * D_;
    for (int i = tid; i < 32 * (D_ / 8); i += 256) {
      const int rr = i >> 7;
      const int cc = (i & 127) * 8;
      *(v8h*)(&Qs[rr][cc]) = *(const v8h*)(src + (size_t)rr * D_ + cc);
    }
  }
  if (tid < 32) { m_st[tid] = -1e30f; l_st[tid] = 0.0f; }
  __syncthreads();

  v8f oacc[2][8];
#pragma unroll
  for (int mi = 0; mi < 2; mi++)
#pragma unroll
    for (int dt = 0; dt < 8; dt++) oacc[mi][dt] = zero_v8f();

  const int ncol = lane & 15;
  const int hi8  = (lane & 16) ? 8 : 0;
#if HAVE_ASYNC_LDS
  const int ck = lane >> 2;              // 0..7 : K row handled by this lane
  const int co = (lane & 3) * 8;         // 0,8,16,24 halves (16B units)
#endif

  for (int j0 = 0; j0 < S_; j0 += 128) {
    // ---- 1. scores for this wave's 16 keys (full-D reduction) ----
    v8f sc[2];
    sc[0] = zero_v8f(); sc[1] = zero_v8f();
    const _Float16* kbase = Kh + ((size_t)(b * S_ + j0 + w * 16)) * D_;
#if HAVE_ASYNC_LDS
    // prologue: async-stage chunk 0 (16 keys x 32 halves = 2 x 512B)
    async_copy16(kbase + (size_t)ck * D_ + co,       &Ksw[w][0][ck][co]);
    async_copy16(kbase + (size_t)(ck + 8) * D_ + co, &Ksw[w][0][ck + 8][co]);
    {
      int idx = 0;
#pragma unroll 8
      for (int kd = 0; kd < D_; kd += 32, idx++) {
        if (kd + 32 < D_) {
          const int nb = (idx + 1) & 1;
          async_copy16(kbase + (size_t)ck * D_ + kd + 32 + co,       &Ksw[w][nb][ck][co]);
          async_copy16(kbase + (size_t)(ck + 8) * D_ + kd + 32 + co, &Ksw[w][nb][ck + 8][co]);
          wait_async_le2();   // current chunk landed; next still in flight
        } else {
          wait_async_0();
        }
        v16h bf = load_frag_b(&Ksw[w][idx & 1][0][0], 40, lane);
        v16h a0 = load_frag_a(&Qs[0][kd],  1032, lane);
        sc[0] = wmma_f16(a0, bf, sc[0]);
        v16h a1 = load_frag_a(&Qs[16][kd], 1032, lane);
        sc[1] = wmma_f16(a1, bf, sc[1]);
      }
    }
#else
#pragma unroll 8
    for (int kd = 0; kd < D_; kd += 32) {
      v16h bf = load_frag_b(kbase + kd, D_, lane);
      v16h a0 = load_frag_a(&Qs[0][kd],  1032, lane);
      sc[0] = wmma_f16(a0, bf, sc[0]);
      v16h a1 = load_frag_a(&Qs[16][kd], 1032, lane);
      sc[1] = wmma_f16(a1, bf, sc[1]);
    }
#endif
    // ---- 2. publish scores ----
#pragma unroll
    for (int mi = 0; mi < 2; mi++)
#pragma unroll
      for (int v = 0; v < 8; v++)
        Sl[mi * 16 + hi8 + v][w * 16 + ncol] = sc[mi][v];
    __syncthreads();

    // ---- 3. online softmax over the 128-key block ----
    {
      const int rrow = tid >> 3;          // 8 threads per row (same wave)
      const int c0   = (tid & 7) * 16;
      float vals[16];
      float mloc = -1e30f;
#pragma unroll
      for (int c = 0; c < 16; c++) {
        vals[c] = Sl[rrow][c0 + c];
        mloc = fmaxf(mloc, vals[c]);
      }
      mloc = fmaxf(mloc, __shfl_xor(mloc, 1, 32));
      mloc = fmaxf(mloc, __shfl_xor(mloc, 2, 32));
      mloc = fmaxf(mloc, __shfl_xor(mloc, 4, 32));
      const float mold = m_st[rrow];
      const float mnew = fmaxf(mold, mloc);
      float ssum = 0.f;
#pragma unroll
      for (int c = 0; c < 16; c++) {
        const float p = __expf(vals[c] - mnew);
        ssum += p;
        Pl[rrow][c0 + c] = (_Float16)p;
      }
      ssum += __shfl_xor(ssum, 1, 32);
      ssum += __shfl_xor(ssum, 2, 32);
      ssum += __shfl_xor(ssum, 4, 32);
      const float scl = __expf(mold - mnew);
      if ((tid & 7) == 0) {
        l_st[rrow]  = l_st[rrow] * scl + ssum;
        m_st[rrow]  = mnew;
        sc_st[rrow] = scl;
      }
    }
    __syncthreads();

    // ---- 4. rescale O, then O += P @ X over this wave's 128 columns ----
    float rs0[8], rs1[8];
#pragma unroll
    for (int v = 0; v < 8; v++) { rs0[v] = sc_st[hi8 + v]; rs1[v] = sc_st[16 + hi8 + v]; }
#pragma unroll
    for (int dt = 0; dt < 8; dt++)
#pragma unroll
      for (int v = 0; v < 8; v++) {
        oacc[0][dt][v] = oacc[0][dt][v] * rs0[v];
        oacc[1][dt][v] = oacc[1][dt][v] * rs1[v];
      }

    const _Float16* xb_base = Xt + (size_t)b * D_ * S_ + (size_t)(w * 128) * S_ + j0;
#pragma unroll
    for (int kk = 0; kk < 4; kk++) {
      v16h pa0 = load_frag_a(&Pl[0][kk * 32],  136, lane);
      v16h pa1 = load_frag_a(&Pl[16][kk * 32], 136, lane);
#pragma unroll
      for (int dt = 0; dt < 8; dt++) {
        v16h xb = load_frag_b(xb_base + (size_t)(dt * 16) * S_ + kk * 32, S_, lane);
        oacc[0][dt] = wmma_f16(pa0, xb, oacc[0][dt]);
        oacc[1][dt] = wmma_f16(pa1, xb, oacc[1][dt]);
      }
    }
  }

  // ---- epilogue: normalize by 1/l and store f32 ----
  float inv0[8], inv1[8];
#pragma unroll
  for (int v = 0; v < 8; v++) {
    inv0[v] = 1.0f / l_st[hi8 + v];
    inv1[v] = 1.0f / l_st[16 + hi8 + v];
  }
  const size_t obase = ((size_t)(b * S_ + q0)) * D_;
#pragma unroll
  for (int dt = 0; dt < 8; dt++) {
    const int d = w * 128 + dt * 16 + ncol;
#pragma unroll
    for (int v = 0; v < 8; v++) {
      out[obase + (size_t)(hi8 + v) * D_ + d]      = oacc[0][dt][v] * inv0[v];
      out[obase + (size_t)(16 + hi8 + v) * D_ + d] = oacc[1][dt][v] * inv1[v];
    }
  }
}

// ---------------------------------------------------------------------------
extern "C" void kernel_launch(void* const* d_in, const int* in_sizes, int n_in,
                              void* d_out, int out_size, void* d_ws, size_t ws_size,
                              hipStream_t stream) {
  (void)in_sizes; (void)n_in; (void)out_size; (void)ws_size;
  const float* x  = (const float*)d_in[0];
  const float* Wq = (const float*)d_in[1];
  const float* bq = (const float*)d_in[2];
  const float* Wk = (const float*)d_in[3];
  const float* bk = (const float*)d_in[4];
  float* out = (float*)d_out;

  _Float16* Qh = (_Float16*)d_ws;
  _Float16* Kh = Qh + (size_t)B_ * S_ * D_;
  _Float16* Xt = Kh + (size_t)B_ * S_ * D_;

  proj_kernel<<<dim3((B_ * S_) / 128, D_ / 128, 2), 256, 0, stream>>>(
      x, Wq, bq, Wk, bk, Qh, Kh);
  transpose_kernel<<<dim3(S_ / 32, D_ / 32, B_), dim3(32, 8), 0, stream>>>(x, Xt);
  attn_kernel<<<dim3(S_ / 32, B_), 256, 0, stream>>>(Qh, Kh, Xt, out);
}
